// GIN_73538430042256
// MI455X (gfx1250) — matrix operand order repaired and checked by reading, not compiled, not run on hardware
//
#include <hip/hip_runtime.h>

#define HID 64
#define WSTRIDE 136   // padded bf16 column stride: 272 B = 68 DWORDs -> no bank conflicts

typedef __attribute__((ext_vector_type(16))) __bf16 v16bf;
typedef __attribute__((ext_vector_type(8)))  __bf16 v8bf;
typedef __attribute__((ext_vector_type(2)))  __bf16 v2bf;
typedef __attribute__((ext_vector_type(8)))  float  v8f;
typedef __attribute__((ext_vector_type(2)))  float  v2f;

struct U32x8 { uint4 lo, hi; };   // 32 bytes, bit-castable to v16bf

__global__ void zero_f32(float* __restrict__ p, long long n) {
  long long i = (long long)blockIdx.x * blockDim.x + threadIdx.x;
  long long stride = (long long)gridDim.x * blockDim.x;
  for (; i < n; i += stride) p[i] = 0.0f;
}

__global__ void degree_kernel(const long long* __restrict__ dst,
                              float* __restrict__ deg, int E) {
  int e = blockIdx.x * blockDim.x + threadIdx.x;
  if (e < E) atomicAdd(&deg[(int)dst[e]], 1.0f);
}

// 16 threads per edge, float4 gather + 4x global_atomic_add_f32 scatter.
__global__ void scatter_kernel(const float* __restrict__ h,
                               const long long* __restrict__ src,
                               const long long* __restrict__ dst,
                               float* __restrict__ agg, int E) {
  long long tid = (long long)blockIdx.x * blockDim.x + threadIdx.x;
  int e = (int)(tid >> 4);
  if (e >= E) return;
  int c = (int)(tid & 15) * 4;
  int s = (int)src[e];
  int d = (int)dst[e];
  const float4 v = *reinterpret_cast<const float4*>(&h[(size_t)s * HID + c]);
  float* a = &agg[(size_t)d * HID + c];
  atomicAdd(a + 0, v.x);
  atomicAdd(a + 1, v.y);
  atomicAdd(a + 2, v.z);
  atomicAdd(a + 3, v.w);
}

// rst = (1+eps)*h + agg/max(deg,1), written in-place into agg.
__global__ void finalize_kernel(const float* __restrict__ h,
                                float* __restrict__ agg,
                                const float* __restrict__ deg,
                                const float* __restrict__ eps, int layer,
                                long long total) {
  long long i = (long long)blockIdx.x * blockDim.x + threadIdx.x;
  if (i >= total) return;
  int node = (int)(i >> 6);  // /HID
  float scale = 1.0f / fmaxf(deg[node], 1.0f);
  float e = eps[layer];
  agg[i] = (1.0f + e) * h[i] + agg[i] * scale;
}

// Y[M x 64] = relu( X[M x K] @ W[K x 64] + bias ), bf16 WMMA with f32 accum.
// One wave computes a 16x64 tile (four 16x16 accumulators). K in {64,128}, %32==0.
// W staged transposed in LDS (bf16) so B fragments load as 2x ds_load_b128.
__global__ void __launch_bounds__(128) gemm_bias_relu_wmma(
    const float* __restrict__ X, const float* __restrict__ W,
    const float* __restrict__ bias, float* __restrict__ Y, int M, int K) {
  __shared__ unsigned short sWt[HID * WSTRIDE];  // bf16, transposed: [col][k], ~17 KB

  // cooperative stage: W[k][n] (f32) -> sWt[n][k] (bf16), packed-pair converts.
  for (int idx = threadIdx.x; idx < (K >> 1) * HID; idx += blockDim.x) {
    int k2 = idx >> 6;      // pair index along K
    int n  = idx & (HID - 1);
    v2f wf;
    wf.x = W[(2 * k2 + 0) * HID + n];
    wf.y = W[(2 * k2 + 1) * HID + n];
    v2bf wb = __builtin_convertvector(wf, v2bf);  // v_cvt_pk_bf16_f32
    *reinterpret_cast<unsigned int*>(&sWt[n * WSTRIDE + 2 * k2]) =
        __builtin_bit_cast(unsigned int, wb);
  }
  __syncthreads();

  int lane = threadIdx.x & 31;
  int wave = threadIdx.x >> 5;
  int Mtiles = (M + 15) >> 4;
  int tileM = blockIdx.x * 4 + wave;
  if (tileM >= Mtiles) return;  // whole wave uniform -> EXEC all-1s preserved

  int half = lane >> 4;   // K half-split (ISA 7.12.2 16-bit layouts)
  int m    = lane & 15;   // A row within tile / B,C,D column within n-block

  v8f acc[4] = {v8f{}, v8f{}, v8f{}, v8f{}};

  int rowA = tileM * 16 + m;
  if (rowA >= M) rowA = M - 1;  // clamp (store is guarded)
  const float* xrow = X + (size_t)rowA * K;

#pragma unroll 2
  for (int k0 = 0; k0 < K; k0 += 32) {
    // Preload all four B fragments (2x ds_load_b128 each) so LDS latency
    // overlaps; keeps the 4 WMMAs back-to-back.
    U32x8 braw[4];
#pragma unroll
    for (int nb = 0; nb < 4; ++nb) {
      int off = (nb * 16 + m) * WSTRIDE + k0 + (half << 3);
      braw[nb].lo = *reinterpret_cast<const uint4*>(&sWt[off]);
      braw[nb].hi = *reinterpret_cast<const uint4*>(&sWt[off + 16]);
    }

    // A fragment: two contiguous 8-float runs per lane (32 B aligned).
    // element i<8  -> K = k0 + 8*half + i
    // element i>=8 -> K = k0 + 16 + 8*half + (i-8)
    v8f xlo = *reinterpret_cast<const v8f*>(xrow + k0 + (half << 3));
    v8f xhi = *reinterpret_cast<const v8f*>(xrow + k0 + 16 + (half << 3));
    if (k0 + 32 < K) __builtin_prefetch(&xrow[k0 + 32], 0, 3);

    v8bf alo = __builtin_convertvector(xlo, v8bf);  // packed v_cvt_pk_bf16_f32
    v8bf ahi = __builtin_convertvector(xhi, v8bf);
    v16bf a = __builtin_shufflevector(alo, ahi, 0, 1, 2, 3, 4, 5, 6, 7,
                                      8, 9, 10, 11, 12, 13, 14, 15);

#pragma unroll
    for (int nb = 0; nb < 4; ++nb) {
      v16bf b = __builtin_bit_cast(v16bf, braw[nb]);
      acc[nb] = __builtin_amdgcn_wmma_f32_16x16x32_bf16(
          false, a, false, b, (short)0, acc[nb], false, false);
    }
  }

  // epilogue: bias + relu + store. C/D layout: lane n-col, VGPR r -> row r+8*half.
#pragma unroll
  for (int nb = 0; nb < 4; ++nb) {
    int col = nb * 16 + m;
    float bv = bias[col];
#pragma unroll
    for (int r = 0; r < 8; ++r) {
      int row = tileM * 16 + half * 8 + r;
      if (row < M) {
        float vv = acc[nb][r] + bv;
        Y[(size_t)row * HID + col] = fmaxf(vv, 0.0f);
      }
    }
  }
}

extern "C" void kernel_launch(void* const* d_in, const int* in_sizes, int n_in,
                              void* d_out, int out_size, void* d_ws, size_t ws_size,
                              hipStream_t stream) {
  const float*     feat = (const float*)d_in[0];
  const long long* src  = (const long long*)d_in[1];
  const long long* dst  = (const long long*)d_in[2];
  const float*     fc_w = (const float*)d_in[3];
  const float*     fc_b = (const float*)d_in[4];
  const float* w[3] = {(const float*)d_in[5], (const float*)d_in[7], (const float*)d_in[9]};
  const float* b[3] = {(const float*)d_in[6], (const float*)d_in[8], (const float*)d_in[10]};
  const float* eps  = (const float*)d_in[11];
  float* out = (float*)d_out;

  const int IN_DIM = 128;
  int Nn = in_sizes[0] / IN_DIM;   // 50000 nodes
  int E  = in_sizes[1];            // 800000 edges

  size_t nh = (size_t)Nn * HID;    // elements per feature buffer
  float* hA  = (float*)d_ws;
  float* hB  = hA + nh;
  float* agg = hB + nh;
  float* deg = agg + nh;

  int Mtiles = (Nn + 15) / 16;
  dim3 gBlk(128);                         // 4 waves / block
  dim3 gGrd((unsigned)((Mtiles + 3) / 4));

  // fc_init + relu: h = relu(features @ fc_w + fc_b), K=128
  gemm_bias_relu_wmma<<<gGrd, gBlk, 0, stream>>>(feat, fc_w, fc_b, hA, Nn, IN_DIM);

  // degree is layer-invariant: compute once per launch
  zero_f32<<<256, 256, 0, stream>>>(deg, (long long)Nn);
  degree_kernel<<<(E + 255) / 256, 256, 0, stream>>>(dst, deg, E);

  float* hcur = hA;
  float* hnext = hB;
  for (int l = 0; l < 3; ++l) {
    zero_f32<<<2048, 256, 0, stream>>>(agg, (long long)nh);
    long long st = (long long)E * 16;
    scatter_kernel<<<(unsigned)((st + 255) / 256), 256, 0, stream>>>(hcur, src, dst, agg, E);
    finalize_kernel<<<(unsigned)((nh + 255) / 256), 256, 0, stream>>>(hcur, agg, deg, eps, l,
                                                                      (long long)nh);
    float* Yl = (l == 2) ? out : hnext;
    gemm_bias_relu_wmma<<<gGrd, gBlk, 0, stream>>>(agg, w[l], b[l], Yl, Nn, HID);
    float* t = hcur; hcur = hnext; hnext = t;
  }
}
